// mLSTMCell_88682484728412
// MI455X (gfx1250) — matrix operand order repaired and checked by reading, not compiled, not run on hardware
//
#include <hip/hip_runtime.h>
#include <math.h>

typedef __attribute__((ext_vector_type(16))) _Float16 v16h;
typedef __attribute__((ext_vector_type(8)))  float    v8f;
typedef __attribute__((ext_vector_type(4)))  float    f4;

#define NB   128      // batch
#define DIN  1024
#define HD   1024

// ---- workspace layout (floats) ----
#define WS_Q   0
#define WS_K   (WS_Q + NB*HD)
#define WS_V   (WS_K + NB*HD)
#define WS_I   (WS_V + NB*HD)
#define WS_F   (WS_I + NB*HD)
#define WS_O   (WS_F + NB*HD)
#define HSPLIT 8
#define WS_HP  (WS_O + NB*HD)                 // [HSPLIT][NB][HD]
#define WS_DEN (WS_HP + HSPLIT*NB*HD)         // [NB]

// ---------------------------------------------------------------------------
// Kernel 1: proj = x @ W_proj + b_proj ; q=tanh, k=tanh, v=copy (g is dead code)
// One wave (32 threads) per 16x16 output tile; K-loop of 32 per WMMA.
// A fragment layout (16-bit A 16x32): lane<16: K={0..7,16..23}; lane>=16: K={8..15,24..31}
// B fragment layout (16-bit B 32x16): elem e of lane L = B[kb + (L>=16?8:0) + (e<8?e:e+8)][n0 + (L&15)]
// C/D layout: VGPR r -> M = r + (L>=16?8:0), N = L&15
// ---------------------------------------------------------------------------
__global__ __launch_bounds__(32)
void gemm_proj_kernel(const float* __restrict__ x, const float* __restrict__ Wp,
                      const float* __restrict__ bp,
                      float* __restrict__ wq, float* __restrict__ wk, float* __restrict__ wv) {
    const int lane = threadIdx.x;
    const int l  = lane & 15;
    const int hi = lane >> 4;
    const int n0 = blockIdx.x * 16;        // 0..3071 (skip dead g columns)
    const int m0 = blockIdx.y * 16;        // 0..127
    const int n  = n0 + l;

    v8f acc = {};
    for (int kb = 0; kb < DIN; kb += 32) {
        const float* arow = x + (size_t)(m0 + l) * DIN + kb + hi * 8;
        v16h a, bm;
        #pragma unroll
        for (int j = 0; j < 8; ++j) {
            a[j]     = (_Float16)arow[j];        // K = kb + hi*8 + j
            a[8 + j] = (_Float16)arow[16 + j];   // K = kb + hi*8 + 16 + j
        }
        const float* bcol = Wp + (size_t)(kb + hi * 8) * (4 * HD) + n;
        #pragma unroll
        for (int e = 0; e < 16; ++e)
            bm[e] = (_Float16)bcol[(size_t)(e < 8 ? e : e + 8) * (4 * HD)];
        acc = __builtin_amdgcn_wmma_f32_16x16x32_f16(false, a, false, bm,
                                                     (short)0, acc, false, false);
    }

    const float bias = bp[n];
    #pragma unroll
    for (int r = 0; r < 8; ++r) {
        const int m = m0 + hi * 8 + r;
        const float val = acc[r] + bias;
        if (n < HD)            wq[(size_t)m * HD + n]            = tanhf(val);
        else if (n < 2 * HD)   wk[(size_t)m * HD + (n - HD)]     = tanhf(val);
        else                   wv[(size_t)m * HD + (n - 2 * HD)] = val;
    }
}

// ---------------------------------------------------------------------------
// Kernel 2: gates = sigmoid(v @ W_gates + b_gates) -> i, f, o
// ---------------------------------------------------------------------------
__global__ __launch_bounds__(32)
void gemm_gates_kernel(const float* __restrict__ wv, const float* __restrict__ Wg,
                       const float* __restrict__ bg,
                       float* __restrict__ wi, float* __restrict__ wf, float* __restrict__ wo) {
    const int lane = threadIdx.x;
    const int l  = lane & 15;
    const int hi = lane >> 4;
    const int n0 = blockIdx.x * 16;        // 0..3071
    const int m0 = blockIdx.y * 16;
    const int n  = n0 + l;

    v8f acc = {};
    for (int kb = 0; kb < HD; kb += 32) {
        const float* arow = wv + (size_t)(m0 + l) * HD + kb + hi * 8;
        v16h a, bm;
        #pragma unroll
        for (int j = 0; j < 8; ++j) {
            a[j]     = (_Float16)arow[j];
            a[8 + j] = (_Float16)arow[16 + j];
        }
        const float* bcol = Wg + (size_t)(kb + hi * 8) * (3 * HD) + n;
        #pragma unroll
        for (int e = 0; e < 16; ++e)
            bm[e] = (_Float16)bcol[(size_t)(e < 8 ? e : e + 8) * (3 * HD)];
        acc = __builtin_amdgcn_wmma_f32_16x16x32_f16(false, a, false, bm,
                                                     (short)0, acc, false, false);
    }

    const float bias = bg[n];
    #pragma unroll
    for (int r = 0; r < 8; ++r) {
        const int m = m0 + hi * 8 + r;
        const float val = 1.0f / (1.0f + __expf(-(acc[r] + bias)));
        if (n < HD)            wi[(size_t)m * HD + n]            = val;
        else if (n < 2 * HD)   wf[(size_t)m * HD + (n - HD)]     = val;
        else                   wo[(size_t)m * HD + (n - 2 * HD)] = val;
    }
}

// ---------------------------------------------------------------------------
// Kernel 3: n_new = f*n + i*k ; denom[b] = max(sum_h q*n_new, 1e-6)
// ---------------------------------------------------------------------------
__global__ __launch_bounds__(256)
void n_denom_kernel(const float* __restrict__ wq, const float* __restrict__ wk,
                    const float* __restrict__ wi, const float* __restrict__ wf,
                    const float* __restrict__ nprev,
                    float* __restrict__ n_out, float* __restrict__ denom) {
    __shared__ float red[256];
    const int b = blockIdx.x, t = threadIdx.x;
    float local = 0.0f;
    for (int h = t; h < HD; h += 256) {
        const size_t idx = (size_t)b * HD + h;
        const float nn = wf[idx] * nprev[idx] + wi[idx] * wk[idx];
        n_out[idx] = nn;
        local += wq[idx] * nn;
    }
    red[t] = local;
    __syncthreads();
    for (int s = 128; s > 0; s >>= 1) {
        if (t < s) red[t] += red[t + s];
        __syncthreads();
    }
    if (t == 0) denom[b] = fmaxf(red[0], 1e-6f);
}

// ---------------------------------------------------------------------------
// Kernel 4 (the bandwidth phase): C_new = f*C + (i*k)*v, fused h_lin partials.
// grid = (HSPLIT, NB); block = 256 threads, each thread owns 4 consecutive d.
// C read + C_new write are nontemporal b128 (1 GB stream; don't pollute L2).
// ---------------------------------------------------------------------------
__global__ __launch_bounds__(256)
void stream_kernel(const float* __restrict__ Cin,
                   const float* __restrict__ wq, const float* __restrict__ wk,
                   const float* __restrict__ wv, const float* __restrict__ wi,
                   const float* __restrict__ wf,
                   float* __restrict__ Cout, float* __restrict__ hpart) {
    __shared__ float qs[HD / HSPLIT], is_[HD / HSPLIT], fs[HD / HSPLIT], ks[HD / HSPLIT];
    const int b     = blockIdx.y;
    const int hbase = blockIdx.x * (HD / HSPLIT);
    const int t     = threadIdx.x;

    if (t < (HD / HSPLIT)) {
        const size_t idx = (size_t)b * HD + hbase + t;
        qs[t]  = wq[idx];
        is_[t] = wi[idx];
        fs[t]  = wf[idx];
        ks[t]  = wk[idx];
    }
    const int d4 = t * 4;
    const f4 v4 = *(const f4*)(wv + (size_t)b * HD + d4);
    __syncthreads();

    const float* cin  = Cin  + (size_t)b * HD * HD + (size_t)hbase * HD + d4;
    float*       cout = Cout + (size_t)b * HD * HD + (size_t)hbase * HD + d4;

    f4 hacc = (f4)(0.0f);
    #pragma unroll 4
    for (int hh = 0; hh < HD / HSPLIT; ++hh) {
        const float fg = fs[hh];
        const float ik = is_[hh] * ks[hh];
        const float qg = qs[hh];
        const f4 c  = __builtin_nontemporal_load((const f4*)cin);
        const f4 cn = fg * c + ik * v4;
        __builtin_nontemporal_store(cn, (f4*)cout);
        hacc += qg * cn;
        cin  += HD;
        cout += HD;
    }
    *(f4*)(hpart + ((size_t)blockIdx.x * NB + b) * HD + d4) = hacc;
}

// ---------------------------------------------------------------------------
// Kernel 5: h_new = o * (sum_p hpart) / denom
// ---------------------------------------------------------------------------
__global__ __launch_bounds__(256)
void finalize_kernel(const float* __restrict__ hpart, const float* __restrict__ wo,
                     const float* __restrict__ denom, float* __restrict__ hout) {
    const int idx = blockIdx.x * 256 + threadIdx.x;   // 0 .. NB*HD-1
    const int b = idx >> 10;
    float s = 0.0f;
    #pragma unroll
    for (int p = 0; p < HSPLIT; ++p)
        s += hpart[((size_t)p * NB + b) * HD + (idx & (HD - 1))];
    hout[idx] = wo[idx] * (s / denom[b]);
}

// ---------------------------------------------------------------------------
extern "C" void kernel_launch(void* const* d_in, const int* in_sizes, int n_in,
                              void* d_out, int out_size, void* d_ws, size_t ws_size,
                              hipStream_t stream) {
    const float* x       = (const float*)d_in[0];   // (128,1024)
    const float* C       = (const float*)d_in[1];   // (128,1024,1024)
    const float* n_state = (const float*)d_in[2];   // (128,1024)
    // d_in[3] = h (unused by reference)
    const float* W_proj  = (const float*)d_in[4];   // (1024,4096)
    const float* b_proj  = (const float*)d_in[5];   // (4096)
    const float* W_gate  = (const float*)d_in[6];   // (1024,3072)
    const float* b_gate  = (const float*)d_in[7];   // (3072)

    float* ws = (float*)d_ws;
    float* wq = ws + WS_Q;
    float* wk = ws + WS_K;
    float* wv = ws + WS_V;
    float* wi = ws + WS_I;
    float* wf = ws + WS_F;
    float* wo = ws + WS_O;
    float* hp = ws + WS_HP;
    float* dn = ws + WS_DEN;

    float* hout = (float*)d_out;                               // (128,1024)
    float* Cout = hout + (size_t)NB * HD;                      // (128,1024,1024)
    float* nout = Cout + (size_t)NB * HD * HD;                 // (128,1024)

    // 1) proj GEMM (q,k,v; g dead-code eliminated): 192 n-tiles x 8 m-tiles
    gemm_proj_kernel<<<dim3(192, 8), 32, 0, stream>>>(x, W_proj, b_proj, wq, wk, wv);
    // 2) gates GEMM (i,f,o)
    gemm_gates_kernel<<<dim3(192, 8), 32, 0, stream>>>(wv, W_gate, b_gate, wi, wf, wo);
    // 3) n_new + denom
    n_denom_kernel<<<NB, 256, 0, stream>>>(wq, wk, wi, wf, n_state, nout, dn);
    // 4) streaming C update fused with h_lin partial reduction
    stream_kernel<<<dim3(HSPLIT, NB), 256, 0, stream>>>(C, wq, wk, wv, wi, wf, Cout, hp);
    // 5) finalize h_new
    finalize_kernel<<<(NB * HD) / 256, 256, 0, stream>>>(hp, wo, dn, hout);
}
